// NST_33758442947077
// MI455X (gfx1250) — matrix operand order repaired and checked by reading, not compiled, not run on hardware
//
#include <hip/hip_runtime.h>
#include <hip/hip_bf16.h>

typedef __attribute__((ext_vector_type(16))) __bf16 v16bf;
typedef __attribute__((ext_vector_type(8)))  float  v8f;

#define B_    64
#define C_    512
#define N_    1024
#define NROWS (B_ * C_)        // 32768 rows per tensor
#define TILE  64
#define KC    64               // bf16 elements staged per K stage
#define PITCH 36               // dwords per LDS row (32 data + 4 pad, 16B aligned)

union FragB { unsigned int u[8]; v16bf v; };

__device__ __forceinline__ unsigned int pack_bf16x2(float a, float b) {
    unsigned int ua = __float_as_uint(a);
    unsigned int ub = __float_as_uint(b);
    ua += 0x7FFFu + ((ua >> 16) & 1u);   // round-to-nearest-even
    ub += 0x7FFFu + ((ub >> 16) & 1u);
    return (ua >> 16) | (ub & 0xFFFF0000u);
}

// ---------------------------------------------------------------------------
// Kernel 1: per-row inverse L2 norms. blockIdx.x in [0, 2*NROWS):
//   [0, NROWS)        -> rows of fm_s -> inv[rid]
//   [NROWS, 2*NROWS)  -> rows of fm_t -> inv[rid]
// ---------------------------------------------------------------------------
__global__ __launch_bounds__(256) void nst_norms(const float* __restrict__ fs,
                                                 const float* __restrict__ ft,
                                                 float* __restrict__ inv) {
    const int rid = blockIdx.x;
    const float* src = (rid < NROWS) ? fs : ft;
    const int r = rid & (NROWS - 1);
    const float4* p = (const float4*)(src + (size_t)r * N_);
    const int t = threadIdx.x;             // 256 threads, 4 floats each
    float4 v = p[t];
    float s = v.x * v.x + v.y * v.y + v.z * v.z + v.w * v.w;
    #pragma unroll
    for (int off = 16; off > 0; off >>= 1) s += __shfl_xor(s, off, 32);
    __shared__ float wsum[8];
    if ((t & 31) == 0) wsum[t >> 5] = s;
    __syncthreads();
    if (t == 0) {
        float tot = 0.f;
        #pragma unroll
        for (int w = 0; w < 8; ++w) tot += wsum[w];
        inv[rid] = 1.0f / fmaxf(sqrtf(tot), 1e-12f);
    }
}

// ---------------------------------------------------------------------------
// Kernel 2: fused batched grams. Block = 64x64 tile (I,J) of batch b.
// Computes partial sum of  G_tt^2 + G_ss^2 - 2*G_ts^2  over its tile.
// ---------------------------------------------------------------------------
__global__ __launch_bounds__(256) void nst_gram(const float* __restrict__ fs,
                                                const float* __restrict__ ft,
                                                const float* __restrict__ invs,
                                                const float* __restrict__ invt,
                                                float* __restrict__ partial) {
    // blk 0 = T_I, 1 = T_J, 2 = S_I, 3 = S_J
    __shared__ __align__(16) unsigned int smem[4][TILE][PITCH];
    __shared__ float scal[4][TILE];
    __shared__ float wsum[8];

    const int t  = threadIdx.x;
    const int jt = blockIdx.x, it = blockIdx.y, b = blockIdx.z;
    const int i0 = it * TILE, j0 = jt * TILE;

    {   // one scale per (blk,row), 256 values total
        const int blk = t >> 6, row = t & 63;
        const int r = ((blk & 1) ? j0 : i0) + row;
        const float* ivp = (blk < 2) ? invt : invs;
        scal[blk][row] = ivp[b * C_ + r];
    }
    __syncthreads();

    const int lane = t & 31;
    const int wave = t >> 5;
    const int ri   = wave & 3;          // A subtile row (0..3)
    const int cjb  = (wave >> 2) * 2;   // first of 2 B subtile cols
    const int hf   = lane >> 4;         // lane half per ISA fragment layout
    const int m    = lane & 15;

    v8f acc_tt[2] = {{}, {}}, acc_ss[2] = {{}, {}}, acc_ts[2] = {{}, {}};

    for (int ks = 0; ks < N_ / KC; ++ks) {
        const int k0 = ks * KC;
        // Stage 4 blocks of 64 rows x 64 floats, scale + convert to bf16
        #pragma unroll
        for (int blk = 0; blk < 4; ++blk) {
            const float* src  = (blk < 2) ? ft : fs;
            const int rbase   = (blk & 1) ? j0 : i0;
            const float* base = src + (size_t)(b * C_ + rbase) * N_ + k0;
            #pragma unroll
            for (int q = 0; q < 4; ++q) {
                const int idx = t + 256 * q;          // 0..1023
                const int row = idx >> 4;
                const int c4  = idx & 15;
                float4 v = *(const float4*)(base + (size_t)row * N_ + c4 * 4);
                const float sc = scal[blk][row];
                unsigned int p0 = pack_bf16x2(v.x * sc, v.y * sc);
                unsigned int p1 = pack_bf16x2(v.z * sc, v.w * sc);
                *(uint2*)&smem[blk][row][c4 * 2] = make_uint2(p0, p1);
            }
        }
        __syncthreads();

        #pragma unroll
        for (int s = 0; s < 2; ++s) {   // two K=32 WMMA steps per stage
            FragB aT, aS, bT[2], bS[2];
            // A 16x32 bf16 layout: VGPR0-3 K=(0..7)+8*hf, VGPR4-7 K=(16..23)+8*hf
            const int aBase = s * 16 + hf * 4;
            const int arow  = ri * 16 + m;
            *(uint4*)&aT.u[0] = *(const uint4*)&smem[0][arow][aBase];
            *(uint4*)&aT.u[4] = *(const uint4*)&smem[0][arow][aBase + 8];
            *(uint4*)&aS.u[0] = *(const uint4*)&smem[2][arow][aBase];
            *(uint4*)&aS.u[4] = *(const uint4*)&smem[2][arow][aBase + 8];
            // B 32x16 bf16 layout: VGPR0-3 K=(0..7)+8*hf, VGPR4-7 K=(16..23)+8*hf
            const int bBase = s * 16 + hf * 4;
            #pragma unroll
            for (int jj = 0; jj < 2; ++jj) {
                const int brow = (cjb + jj) * 16 + m;
                *(uint4*)&bT[jj].u[0] = *(const uint4*)&smem[1][brow][bBase];
                *(uint4*)&bT[jj].u[4] = *(const uint4*)&smem[1][brow][bBase + 8];
                *(uint4*)&bS[jj].u[0] = *(const uint4*)&smem[3][brow][bBase];
                *(uint4*)&bS[jj].u[4] = *(const uint4*)&smem[3][brow][bBase + 8];
            }
            #pragma unroll
            for (int jj = 0; jj < 2; ++jj) {
                acc_tt[jj] = __builtin_amdgcn_wmma_f32_16x16x32_bf16(
                    false, aT.v, false, bT[jj].v, (short)0, acc_tt[jj], false, false);
                acc_ss[jj] = __builtin_amdgcn_wmma_f32_16x16x32_bf16(
                    false, aS.v, false, bS[jj].v, (short)0, acc_ss[jj], false, false);
                acc_ts[jj] = __builtin_amdgcn_wmma_f32_16x16x32_bf16(
                    false, aT.v, false, bS[jj].v, (short)0, acc_ts[jj], false, false);
            }
        }
        __syncthreads();
    }

    // per-lane squared combination, then wave32 + block reduction
    float p = 0.f;
    #pragma unroll
    for (int jj = 0; jj < 2; ++jj)
        #pragma unroll
        for (int e = 0; e < 8; ++e) {
            p += acc_tt[jj][e] * acc_tt[jj][e];
            p += acc_ss[jj][e] * acc_ss[jj][e];
            p -= 2.0f * acc_ts[jj][e] * acc_ts[jj][e];
        }
    #pragma unroll
    for (int off = 16; off > 0; off >>= 1) p += __shfl_xor(p, off, 32);
    if (lane == 0) wsum[wave] = p;
    __syncthreads();
    if (t == 0) {
        float tot = 0.f;
        #pragma unroll
        for (int w = 0; w < 8; ++w) tot += wsum[w];
        partial[(b * 8 + it) * 8 + jt] = tot;
    }
}

// ---------------------------------------------------------------------------
// Kernel 3: deterministic fixed-order reduction of 4096 partials -> scalar
// ---------------------------------------------------------------------------
__global__ __launch_bounds__(256) void nst_finalize(const float* __restrict__ partial,
                                                    float* __restrict__ out) {
    __shared__ float red[256];
    const int t = threadIdx.x;
    float s = 0.f;
    for (int i = t; i < 4096; i += 256) s += partial[i];
    red[t] = s;
    __syncthreads();
    for (int k = 128; k > 0; k >>= 1) {
        if (t < k) red[t] += red[t + k];
        __syncthreads();
    }
    if (t == 0) out[0] = red[0] * (1.0f / 16777216.0f);  // LAMB / (B*C*C)
}

extern "C" void kernel_launch(void* const* d_in, const int* in_sizes, int n_in,
                              void* d_out, int out_size, void* d_ws, size_t ws_size,
                              hipStream_t stream) {
    const float* fs = (const float*)d_in[0];   // fm_s
    const float* ft = (const float*)d_in[1];   // fm_t
    float* ws      = (float*)d_ws;
    float* invs    = ws;                 // NROWS floats
    float* invt    = ws + NROWS;         // NROWS floats
    float* partial = ws + 2 * NROWS;     // 4096 floats

    nst_norms<<<dim3(2 * NROWS), dim3(256), 0, stream>>>(fs, ft, ws);
    nst_gram <<<dim3(8, 8, B_),  dim3(256), 0, stream>>>(fs, ft, invs, invt, partial);
    nst_finalize<<<dim3(1), dim3(256), 0, stream>>>(partial, (float*)d_out);
}
